// MIL_Cluster_FC_surv_18605798326623
// MI455X (gfx1250) — matrix operand order repaired and checked by reading, not compiled, not run on hardware
//
#include <hip/hip_runtime.h>
#include <hip/hip_bf16.h>

// MIL cluster-FC survival model, CDNA5 (gfx1250) implementation.
// Dominant GEMMs run on v_wmma_f32_16x16x32_bf16 (f32 accum).
// Next K-slice is warmed with global_prefetch_b8 (__builtin_prefetch) — zero
// register cost (the register-staged variant spilled to scratch).
// Workspace layout (requires ws_size >= 8 MiB):
//   [0]        counts      4 x int
//   [4096]     sums        4*512 x f32 (segment sums)
//   [16384]    idx         4*32768 x int (compacted per-cluster row lists)
//   [1 MiB]    W1t bf16    [C][H][F_IN]  (transposed, 4 MiB)
//   [6 MiB]    W2t bf16    [C][H][H]     (transposed, 2 MiB)

#define NROWS 32768
#define NC    4
#define FIN   1024
#define HH    512
#define DD    256
#define TILE_M 32

typedef __attribute__((ext_vector_type(16))) __bf16 v16bf;
typedef __attribute__((ext_vector_type(8)))  float  v8f;

union FragAB { v16bf v; unsigned int u[8]; };
union FragCD { v8f v; float f[8]; };

__global__ __launch_bounds__(256)
void init_ws_kernel(int* counts, float* sums) {
    int i = blockIdx.x * blockDim.x + threadIdx.x;
    if (i < NC) counts[i] = 0;
    if (i < NC * HH) sums[i] = 0.f;
}

__global__ __launch_bounds__(256)
void convert_weights_kernel(const float* __restrict__ W1, const float* __restrict__ W2,
                            __bf16* __restrict__ W1t, __bf16* __restrict__ W2t) {
    const int n1 = NC * HH * FIN;   // 2M elems
    const int n2 = NC * HH * HH;    // 1M elems
    const int stride = gridDim.x * blockDim.x;
    for (int i = blockIdx.x * blockDim.x + threadIdx.x; i < n1; i += stride) {
        int k = i % FIN; int t = i / FIN; int n = t % HH; int c = t / HH;
        W1t[i] = (__bf16)W1[((size_t)c * FIN + k) * HH + n];   // [c][k][n] -> [c][n][k]
    }
    for (int i = blockIdx.x * blockDim.x + threadIdx.x; i < n2; i += stride) {
        int k = i % HH; int t = i / HH; int n = t % HH; int c = t / HH;
        W2t[i] = (__bf16)W2[((size_t)c * HH + k) * HH + n];
    }
}

__global__ __launch_bounds__(256)
void partition_rows_kernel(const int* __restrict__ cid, int* counts, int* idx) {
    const int stride = gridDim.x * blockDim.x;
    for (int i = blockIdx.x * blockDim.x + threadIdx.x; i < NROWS; i += stride) {
        int c = cid[i];
        if (c >= 0 && c < NC) {
            int p = atomicAdd(&counts[c], 1);
            idx[c * NROWS + p] = i;
        }
    }
}

// 8 waves/block: wave = (mt, nq); each wave: 16 rows x 128 cols = 8 WMMA accums.
__global__ __launch_bounds__(256)
void mlp_cluster_kernel(const float* __restrict__ x,
                        const int* __restrict__ idx,
                        const int* __restrict__ counts,
                        const __bf16* __restrict__ W1t,
                        const __bf16* __restrict__ W2t,
                        const float* __restrict__ b1,
                        const float* __restrict__ b2,
                        float* __restrict__ sums) {
    __shared__ __align__(16) __bf16 Bs[HH * 32];       // 32 KB: B slice, n-major [n][32]
    __shared__ __align__(16) __bf16 H1s[TILE_M * HH];  // 32 KB: h1 (A-slice aliases start)

    const int c    = blockIdx.x >> 10;
    const int tile = blockIdx.x & 1023;
    const int cnt  = counts[c];
    const int rowbase = tile * TILE_M;
    if (rowbase >= cnt) return;
    int nvalid = cnt - rowbase; if (nvalid > TILE_M) nvalid = TILE_M;

    const int tid  = threadIdx.x;
    const int wave = tid >> 5;
    const int lane = tid & 31;
    const int lo   = lane & 15;
    const int hi   = lane >> 4;
    const int mt   = wave & 1;   // row tile (16 rows)
    const int nq   = wave >> 1;  // 128-col group

    const int* idxc = idx + c * NROWS;
    const __bf16* W1c = W1t + (size_t)c * HH * FIN;
    const __bf16* W2c = W2t + (size_t)c * HH * HH;
    __bf16* As = H1s;  // 32x32 A slice aliases h1 region during stage 1

    // gather index for A-slice loading (thread -> row tid/8, 4 cols)
    const int arow = tid >> 3;
    const int acg  = (tid & 7) << 2;
    const int gidx = (arow < nvalid) ? idxc[rowbase + arow] : -1;

    // ---------------- Stage 1: h1 = relu(x @ W1 + b1), K = 1024 ----------------
    FragCD acc[8];
#pragma unroll
    for (int nt = 0; nt < 8; ++nt)
#pragma unroll
        for (int i = 0; i < 8; ++i) acc[nt].f[i] = 0.f;

    for (int k0 = 0; k0 < FIN; k0 += 32) {
        __syncthreads();
        // A slice 32x32 bf16 (invalid rows -> 0)
        float4 xv = {0.f, 0.f, 0.f, 0.f};
        if (gidx >= 0) xv = *(const float4*)(x + (size_t)gidx * FIN + k0 + acg);
        __bf16* ad = As + arow * 32 + acg;
        ad[0] = (__bf16)xv.x; ad[1] = (__bf16)xv.y;
        ad[2] = (__bf16)xv.z; ad[3] = (__bf16)xv.w;
        // B slice 512x32 bf16, n-major
#pragma unroll
        for (int r = 0; r < 2; ++r) {
            int n = (tid << 1) + r;
            const uint4* src = (const uint4*)(W1c + (size_t)n * FIN + k0);
            uint4* dst = (uint4*)(Bs + n * 32);
            dst[0] = src[0]; dst[1] = src[1]; dst[2] = src[2]; dst[3] = src[3];
        }
        __syncthreads();
        // warm the next K-slice (global_prefetch_b8; no registers, no counters)
        if (k0 + 32 < FIN) {
            __builtin_prefetch(W1c + (size_t)(tid << 1) * FIN + k0 + 32, 0, 0);
            __builtin_prefetch(W1c + (size_t)((tid << 1) + 1) * FIN + k0 + 32, 0, 0);
            if (gidx >= 0)
                __builtin_prefetch(x + (size_t)gidx * FIN + k0 + 32 + acg, 0, 0);
        }
        // A fragment: lane lo = M row, hi selects K+8; pairs contiguous
        FragAB a;
        const __bf16* ap = As + (mt * 16 + lo) * 32 + 8 * hi;
#pragma unroll
        for (int v = 0; v < 8; ++v) {
            int k = (v < 4) ? (2 * v) : (8 + 2 * v);
            a.u[v] = *(const unsigned int*)(ap + k);
        }
#pragma unroll
        for (int nt = 0; nt < 8; ++nt) {
            FragAB b;
            const __bf16* bp = Bs + (nq * 128 + nt * 16 + lo) * 32 + 8 * hi;
#pragma unroll
            for (int v = 0; v < 8; ++v) b.u[v] = *(const unsigned int*)(bp + ((v < 4) ? (2 * v) : (8 + 2 * v)));
            acc[nt].v = __builtin_amdgcn_wmma_f32_16x16x32_bf16(
                false, a.v, false, b.v, (short)0, acc[nt].v, false, false);
        }
    }
    __syncthreads();
    // write h1 (relu + bias) to LDS as bf16; C/D layout: row = r + 8*hi
    const float* b1c = b1 + c * HH;
#pragma unroll
    for (int nt = 0; nt < 8; ++nt) {
        int nc = nq * 128 + nt * 16 + lo;
        float bias = b1c[nc];
#pragma unroll
        for (int r = 0; r < 8; ++r) {
            int m = mt * 16 + 8 * hi + r;
            float vv = acc[nt].f[r] + bias;
            H1s[m * HH + nc] = (__bf16)(vv > 0.f ? vv : 0.f);
        }
    }
    __syncthreads();

    // ---------------- Stage 2: y = relu(h1 @ W2 + b2), K = 512 ----------------
    FragCD acc2[8];
#pragma unroll
    for (int nt = 0; nt < 8; ++nt)
#pragma unroll
        for (int i = 0; i < 8; ++i) acc2[nt].f[i] = 0.f;

    for (int k0 = 0; k0 < HH; k0 += 32) {
        __syncthreads();
#pragma unroll
        for (int r = 0; r < 2; ++r) {
            int n = (tid << 1) + r;
            const uint4* src = (const uint4*)(W2c + (size_t)n * HH + k0);
            uint4* dst = (uint4*)(Bs + n * 32);
            dst[0] = src[0]; dst[1] = src[1]; dst[2] = src[2]; dst[3] = src[3];
        }
        __syncthreads();
        if (k0 + 32 < HH) {
            __builtin_prefetch(W2c + (size_t)(tid << 1) * HH + k0 + 32, 0, 0);
            __builtin_prefetch(W2c + (size_t)((tid << 1) + 1) * HH + k0 + 32, 0, 0);
        }
        FragAB a;
        const __bf16* ap = H1s + (mt * 16 + lo) * HH + k0 + 8 * hi;
#pragma unroll
        for (int v = 0; v < 8; ++v) {
            int k = (v < 4) ? (2 * v) : (8 + 2 * v);
            a.u[v] = *(const unsigned int*)(ap + k);
        }
#pragma unroll
        for (int nt = 0; nt < 8; ++nt) {
            FragAB b;
            const __bf16* bp = Bs + (nq * 128 + nt * 16 + lo) * 32 + 8 * hi;
#pragma unroll
            for (int v = 0; v < 8; ++v) b.u[v] = *(const unsigned int*)(bp + ((v < 4) ? (2 * v) : (8 + 2 * v)));
            acc2[nt].v = __builtin_amdgcn_wmma_f32_16x16x32_bf16(
                false, a.v, false, b.v, (short)0, acc2[nt].v, false, false);
        }
    }
    __syncthreads();

    // ---------------- Column reduction into per-cluster sums ----------------
    float* colsum = (float*)Bs;  // 512 floats, reuse B slice LDS
    colsum[tid] = 0.f; colsum[tid + 256] = 0.f;
    __syncthreads();
    const float* b2c = b2 + c * HH;
#pragma unroll
    for (int nt = 0; nt < 8; ++nt) {
        int nc = nq * 128 + nt * 16 + lo;
        float bias = b2c[nc];
        float p = 0.f;
#pragma unroll
        for (int r = 0; r < 8; ++r) {
            int m = mt * 16 + 8 * hi + r;
            if (m < nvalid) {
                float vv = acc2[nt].f[r] + bias;
                p += (vv > 0.f ? vv : 0.f);
            }
        }
        atomicAdd(&colsum[nc], p);
    }
    __syncthreads();
    atomicAdd(&sums[c * HH + tid], colsum[tid]);
    atomicAdd(&sums[c * HH + tid + 256], colsum[tid + 256]);
}

// Tiny head: means -> fc -> gated attention -> softmax -> pooled -> rho -> logits.
__global__ __launch_bounds__(256)
void epilogue_kernel(const float* __restrict__ sums, const int* __restrict__ counts,
                     const float* __restrict__ fcW, const float* __restrict__ fcB,
                     const float* __restrict__ aW,  const float* __restrict__ aB,
                     const float* __restrict__ gW,  const float* __restrict__ gB,
                     const float* __restrict__ cW,  const float* __restrict__ cB,
                     const float* __restrict__ rhoW, const float* __restrict__ rhoB,
                     const float* __restrict__ clsW, const float* __restrict__ clsB,
                     float* __restrict__ out) {
    __shared__ float hM[NC * HH];
    __shared__ float hX[NC * HH];
    __shared__ float ag[NC * DD];
    __shared__ float sc[NC];
    __shared__ float hp[HH];
    __shared__ float dp[DD];
    __shared__ float lg[NC];
    const int tid = threadIdx.x;

    for (int i = tid; i < NC * HH; i += 256) {
        int c = i >> 9;
        float cnt = (float)counts[c]; if (cnt < 1.f) cnt = 1.f;
        hM[i] = sums[i] / cnt;
    }
    __syncthreads();
    for (int i = tid; i < NC * HH; i += 256) {           // h = relu(means @ fcW + fcB)
        int c = i >> 9, j = i & 511;
        float s = fcB[j];
        for (int k = 0; k < HH; ++k) s += hM[c * HH + k] * fcW[k * HH + j];
        hX[i] = s > 0.f ? s : 0.f;
    }
    __syncthreads();
    for (int i = tid; i < NC * DD; i += 256) {           // a*g gated attention
        int c = i >> 8, j = i & 255;
        float sa = aB[j], sg = gB[j];
        for (int k = 0; k < HH; ++k) {
            float h = hX[c * HH + k];
            sa += h * aW[k * DD + j];
            sg += h * gW[k * DD + j];
        }
        ag[i] = tanhf(sa) * (1.f / (1.f + expf(-sg)));
    }
    __syncthreads();
    if (tid < NC) {                                       // A = (a*g) @ cW + cB
        float s = cB[0];
        for (int k = 0; k < DD; ++k) s += ag[tid * DD + k] * cW[k];
        sc[tid] = s;
    }
    __syncthreads();
    if (tid == 0) {                                       // softmax over 4
        float mx = sc[0];
        for (int c = 1; c < NC; ++c) mx = sc[c] > mx ? sc[c] : mx;
        float den = 0.f;
        for (int c = 0; c < NC; ++c) { float e = expf(sc[c] - mx); sc[c] = e; den += e; }
        for (int c = 0; c < NC; ++c) sc[c] /= den;
    }
    __syncthreads();
    for (int j = tid; j < HH; j += 256) {                 // h_path = A @ h
        float s = 0.f;
        for (int c = 0; c < NC; ++c) s += sc[c] * hX[c * HH + j];
        hp[j] = s;
    }
    __syncthreads();
    if (tid < DD) {                                       // relu(h_path @ rhoW + rhoB)
        float s = rhoB[tid];
        for (int k = 0; k < HH; ++k) s += hp[k] * rhoW[k * DD + tid];
        dp[tid] = s > 0.f ? s : 0.f;
    }
    __syncthreads();
    if (tid < NC) {                                       // logits
        float s = clsB[tid];
        for (int k = 0; k < DD; ++k) s += dp[k] * clsW[k * NC + tid];
        lg[tid] = s;
    }
    __syncthreads();
    if (tid == 0) {
        float S = 1.f; int am = 0; float best = lg[0];
        for (int l = 0; l < NC; ++l) {
            float hz = 1.f / (1.f + expf(-lg[l]));
            out[l] = hz;
            S *= (1.f - hz);
            out[NC + l] = S;
            if (lg[l] > best) { best = lg[l]; am = l; }
        }
        out[2 * NC] = (float)am;
    }
}

extern "C" void kernel_launch(void* const* d_in, const int* in_sizes, int n_in,
                              void* d_out, int out_size, void* d_ws, size_t ws_size,
                              hipStream_t stream) {
    const float* x    = (const float*)d_in[0];
    const int*   cid  = (const int*)d_in[1];
    const float* W1   = (const float*)d_in[2];
    const float* b1   = (const float*)d_in[3];
    const float* W2   = (const float*)d_in[4];
    const float* b2   = (const float*)d_in[5];
    const float* fcW  = (const float*)d_in[6];
    const float* fcB  = (const float*)d_in[7];
    const float* aW   = (const float*)d_in[8];
    const float* aB   = (const float*)d_in[9];
    const float* gW   = (const float*)d_in[10];
    const float* gB   = (const float*)d_in[11];
    const float* cW   = (const float*)d_in[12];
    const float* cB   = (const float*)d_in[13];
    const float* rhoW = (const float*)d_in[14];
    const float* rhoB = (const float*)d_in[15];
    const float* clsW = (const float*)d_in[16];
    const float* clsB = (const float*)d_in[17];

    char* ws = (char*)d_ws;
    int*    counts = (int*)(ws);
    float*  sums   = (float*)(ws + 4096);
    int*    idx    = (int*)(ws + 16384);
    __bf16* W1t    = (__bf16*)(ws + (1u << 20));
    __bf16* W2t    = (__bf16*)(ws + (6u << 20));

    init_ws_kernel<<<8, 256, 0, stream>>>(counts, sums);
    convert_weights_kernel<<<4096, 256, 0, stream>>>(W1, W2, W1t, W2t);
    partition_rows_kernel<<<128, 256, 0, stream>>>(cid, counts, idx);
    mlp_cluster_kernel<<<NC * (NROWS / TILE_M), 256, 0, stream>>>(
        x, idx, counts, W1t, W2t, b1, b2, sums);
    epilogue_kernel<<<1, 256, 0, stream>>>(sums, counts, fcW, fcB, aW, aB,
                                           gW, gB, cW, cB, rhoW, rhoB,
                                           clsW, clsB, (float*)d_out);
}
